// AGNN_16286515986689
// MI455X (gfx1250) — matrix operand order repaired and checked by reading, not compiled, not run on hardware
//
#include <hip/hip_runtime.h>
#include <hip/hip_bf16.h>

// ---------------------------------------------------------------------------
// AGNN (2-layer MLP + 2 AGNN propagation passes) for MI455X / gfx1250.
//  - Dense GEMMs: v_wmma_f32_16x16x32_bf16 with bf16x3 fp32-emulation split
//    (hi*Whi + hi*Wlo + lo*Whi, f32 accumulate). W staged in LDS pre-split
//    AND pre-swizzled into WMMA fragment order -> B operand = 2x ds_load_b128.
//  - Edge phase: 4 lanes per edge, contiguous 64B per lane (quad covers the
//    full 256B feature row), shfl-reduced dot; scatter-softmax as
//    unnormalized accumulate + per-node divide. h is L2-resident (25.6MB).
// ---------------------------------------------------------------------------

typedef __attribute__((ext_vector_type(16))) __bf16 v16bf;
typedef __attribute__((ext_vector_type(8)))  float  v8f;

#define IN_C  128
#define HID_C 64
#define OUT_C 40

// ---------------------------- WMMA GEMM ------------------------------------
// Y[nrows, NCOLS] = act(X[nrows, K] @ W[K, NCOLS] + bias)
// One wave computes a 16-row stripe across NT column tiles of 16.
template <int K, int NT, int NCOLS, bool RELU>
__global__ __launch_bounds__(256)
void gemm_bf16x3(const float* __restrict__ X, const float* __restrict__ W,
                 const float* __restrict__ bias, float* __restrict__ Y,
                 int nrows) {
  constexpr int KSTEPS = K / 32;
  constexpr int NFRAG  = KSTEPS * NT;          // B fragments (32 lanes x 16 bf16)
  // Fragment-order LDS: frag-major, lane-major, 16 contiguous bf16 per lane.
  __shared__ __bf16 sWhi[NFRAG * 32 * 16];
  __shared__ __bf16 sWlo[NFRAG * 32 * 16];

  // Cooperative load + bf16 hi/lo split + swizzle into WMMA B-fragment order.
  // B layout (ISA 7.12.2): lane = column (lanes 0-15 K={0..7,16..23},
  // lanes 16-31 K={8..15,24..31}), 16 bf16 per lane.
  for (int idx = threadIdx.x; idx < NFRAG * 32 * 16; idx += 256) {
    const int t    = idx & 15;
    const int l    = (idx >> 4) & 31;
    const int frag = idx >> 9;                 // frag = ks*NT + ct
    const int ks   = frag / NT;
    const int ct   = frag % NT;
    const int k    = ks * 32 + (t >> 3) * 16 + (l >> 4) * 8 + (t & 7);
    const int col  = (l & 15) + ct * 16;
    const float v  = (col < NCOLS) ? W[k * NCOLS + col] : 0.0f;
    const __bf16 hi = (__bf16)v;
    sWhi[idx] = hi;
    sWlo[idx] = (__bf16)(v - (float)hi);
  }
  __syncthreads();

  const int wid     = blockIdx.x * 8 + (threadIdx.x >> 5);
  const int lane    = threadIdx.x & 31;
  const int rowBase = wid * 16;
  if (rowBase >= nrows) return;                // wave-uniform: EXEC all-1s

  const int  half  = lane >> 4;
  const int  m     = lane & 15;
  const int  row   = rowBase + m;
  const bool rowOK = (row < nrows);
  const float4* xrow4 = (const float4*)(X + (size_t)row * K);

  v8f acc[NT];
#pragma unroll
  for (int t = 0; t < NT; ++t)
#pragma unroll
    for (int r = 0; r < 8; ++r) acc[t][r] = 0.0f;

#pragma unroll
  for (int ks = 0; ks < KSTEPS; ++ks) {
    const int k0 = ks * 32;
    // A fragment (16x32 bf16): lanes 0-15 hold K {0..7,16..23}, lanes 16-31
    // hold K {8..15,24..31}; row = lane&15.  Two contiguous 8-float runs.
    const int ka = k0 + half * 8;              // multiple of 8 floats
    const int kb = k0 + 16 + half * 8;
    float4 f0, f1, f2, f3;
    if (rowOK) {
      f0 = xrow4[(ka >> 2) + 0];
      f1 = xrow4[(ka >> 2) + 1];
      f2 = xrow4[(kb >> 2) + 0];
      f3 = xrow4[(kb >> 2) + 1];
    } else {
      f0 = f1 = f2 = f3 = make_float4(0.f, 0.f, 0.f, 0.f);
    }
    const float fv[16] = {f0.x, f0.y, f0.z, f0.w, f1.x, f1.y, f1.z, f1.w,
                          f2.x, f2.y, f2.z, f2.w, f3.x, f3.y, f3.z, f3.w};
    v16bf ahi, alo;
#pragma unroll
    for (int t = 0; t < 16; ++t) {
      const __bf16 h = (__bf16)fv[t];
      ahi[t] = h;
      alo[t] = (__bf16)(fv[t] - (float)h);
    }

#pragma unroll
    for (int ct = 0; ct < NT; ++ct) {
      const int off = ((ks * NT + ct) * 32 + lane) * 16;  // 32B-aligned
      const v16bf bhi = *(const v16bf*)(sWhi + off);      // 2x ds_load_b128
      const v16bf blo = *(const v16bf*)(sWlo + off);
      acc[ct] = __builtin_amdgcn_wmma_f32_16x16x32_bf16(
          false, ahi, false, bhi, (short)0, acc[ct], false, false);
      acc[ct] = __builtin_amdgcn_wmma_f32_16x16x32_bf16(
          false, ahi, false, blo, (short)0, acc[ct], false, false);
      acc[ct] = __builtin_amdgcn_wmma_f32_16x16x32_bf16(
          false, alo, false, bhi, (short)0, acc[ct], false, false);
    }
  }

  // C/D layout: VGPR r -> M = r + half*8, N = lane&15.
  const int nOut = lane & 15;
#pragma unroll
  for (int ct = 0; ct < NT; ++ct) {
    const int col = ct * 16 + nOut;
    if (col >= NCOLS) continue;
    const float bb = bias[col];
#pragma unroll
    for (int r = 0; r < 8; ++r) {
      const int orow = rowBase + half * 8 + r;
      if (orow < nrows) {
        float v = acc[ct][r] + bb;
        if (RELU) v = fmaxf(v, 0.0f);
        Y[(size_t)orow * NCOLS + col] = v;
      }
    }
  }
}

// --------------------------- node kernels ----------------------------------
// Quad-per-node: 4 lanes each own a contiguous 64B quarter of the 256B row.

__global__ void invnorm_kernel(const float* __restrict__ h,
                               float* __restrict__ inv, int n) {
  const int gtid = blockIdx.x * blockDim.x + threadIdx.x;
  const int v = gtid >> 2, sub = gtid & 3;
  if (v >= n) return;
  const float4* p = (const float4*)(h + (size_t)v * HID_C) + sub * 4;
  float ss = 0.0f;
#pragma unroll
  for (int i = 0; i < 4; ++i) {
    float4 a = p[i];
    ss += a.x * a.x + a.y * a.y + a.z * a.z + a.w * a.w;
  }
  ss += __shfl_xor(ss, 1);
  ss += __shfl_xor(ss, 2);
  if (sub == 0) inv[v] = rsqrtf(fmaxf(ss, 1e-24f));
}

__global__ void init_kernel(float* __restrict__ amax, float* __restrict__ denom,
                            float* __restrict__ agg, int n) {
  const int i = blockIdx.x * blockDim.x + threadIdx.x;
  if (i < n * HID_C) agg[i] = 0.0f;
  if (i < n) {
    amax[i]  = -__builtin_inff();
    denom[i] = 0.0f;
  }
}

// agg[v] /= denom[v]; optionally emit inverse L2 norm of the result.
__global__ void finalize_kernel(float* __restrict__ agg,
                                const float* __restrict__ denom, int n,
                                float* __restrict__ invOut) {
  const int gtid = blockIdx.x * blockDim.x + threadIdx.x;
  const int v = gtid >> 2, sub = gtid & 3;
  if (v >= n) return;
  const float r = 1.0f / denom[v];
  float* row = agg + (size_t)v * HID_C + sub * 16;
  float ss = 0.0f;
#pragma unroll
  for (int i = 0; i < 16; ++i) {
    const float x = row[i] * r;
    row[i] = x;
    ss += x * x;
  }
  if (invOut) {
    ss += __shfl_xor(ss, 1);
    ss += __shfl_xor(ss, 2);
    if (sub == 0) invOut[v] = rsqrtf(fmaxf(ss, 1e-24f));
  }
}

// --------------------------- edge kernels ----------------------------------

__device__ __forceinline__ void atomicMaxF(float* addr, float v) {
  // Monotonic-bit-pattern trick; valid mixed because int-max only raises
  // toward positives and uint-min only raises among negatives.
  if (v >= 0.0f) atomicMax((int*)addr, __float_as_int(v));
  else           atomicMin((unsigned int*)addr, __float_as_uint(v));
}

// Quad-per-edge: 4 lanes split the 64-channel dot, reduce via shfl_xor.
__global__ void edge_alpha_kernel(const float* __restrict__ h,
                                  const float* __restrict__ inv,
                                  const long long* __restrict__ src,
                                  const long long* __restrict__ dst,
                                  int E, int N,
                                  const float* __restrict__ betaPtr,
                                  float* __restrict__ alpha,
                                  float* __restrict__ amax) {
  const int gtid = blockIdx.x * blockDim.x + threadIdx.x;
  const int e = gtid >> 2, sub = gtid & 3;
  if (e >= E + N) return;
  int s, d;
  if (e < E) { s = (int)src[e]; d = (int)dst[e]; } else { s = d = e - E; }
  const float4* ps = (const float4*)(h + (size_t)s * HID_C) + sub * 4;
  const float4* pd = (const float4*)(h + (size_t)d * HID_C) + sub * 4;
  float dot = 0.0f;
#pragma unroll
  for (int i = 0; i < 4; ++i) {
    float4 a = ps[i];
    float4 b = pd[i];
    dot += a.x * b.x + a.y * b.y + a.z * b.z + a.w * b.w;
  }
  dot += __shfl_xor(dot, 1);
  dot += __shfl_xor(dot, 2);
  if (sub == 0) {
    const float beta = betaPtr ? betaPtr[0] : 1.0f;
    const float al = beta * dot * inv[s] * inv[d];
    alpha[e] = al;
    atomicMaxF(amax + d, al);
  }
}

// Quad-per-edge: each lane scatters 16 contiguous channels.
__global__ void edge_accum_kernel(const float* __restrict__ h,
                                  const float* __restrict__ alpha,
                                  const float* __restrict__ amax,
                                  const long long* __restrict__ src,
                                  const long long* __restrict__ dst,
                                  int E, int N,
                                  float* __restrict__ denom,
                                  float* __restrict__ agg) {
  const int gtid = blockIdx.x * blockDim.x + threadIdx.x;
  const int e = gtid >> 2, sub = gtid & 3;
  if (e >= E + N) return;
  int s, d;
  if (e < E) { s = (int)src[e]; d = (int)dst[e]; } else { s = d = e - E; }
  const float w = __expf(alpha[e] - amax[d]);
  if (sub == 0) atomicAdd(denom + d, w);
  const float4* ps = (const float4*)(h + (size_t)s * HID_C) + sub * 4;
  float* out = agg + (size_t)d * HID_C + sub * 16;
#pragma unroll
  for (int i = 0; i < 4; ++i) {
    float4 a = ps[i];
    atomicAdd(out + 4 * i + 0, a.x * w);
    atomicAdd(out + 4 * i + 1, a.y * w);
    atomicAdd(out + 4 * i + 2, a.z * w);
    atomicAdd(out + 4 * i + 3, a.w * w);
  }
}

// --------------------------- epilogue --------------------------------------

__global__ void logsoftmax_kernel(const float* __restrict__ logits,
                                  float* __restrict__ out, int n) {
  const int v = blockIdx.x * blockDim.x + threadIdx.x;
  if (v >= n) return;
  const float* row = logits + (size_t)v * OUT_C;
  float m = -3.402823466e38f;
#pragma unroll
  for (int i = 0; i < OUT_C; ++i) m = fmaxf(m, row[i]);
  float s = 0.0f;
#pragma unroll
  for (int i = 0; i < OUT_C; ++i) s += __expf(row[i] - m);
  const float ls = __logf(s);
  float* o = out + (size_t)v * OUT_C;
#pragma unroll
  for (int i = 0; i < OUT_C; ++i) o[i] = row[i] - m - ls;
}

// --------------------------- launcher --------------------------------------

extern "C" void kernel_launch(void* const* d_in, const int* in_sizes, int n_in,
                              void* d_out, int out_size, void* d_ws,
                              size_t ws_size, hipStream_t stream) {
  const float*     x     = (const float*)d_in[0];
  const long long* ei    = (const long long*)d_in[1];
  const float*     W1    = (const float*)d_in[2];
  const float*     b1    = (const float*)d_in[3];
  const float*     W2    = (const float*)d_in[4];
  const float*     b2    = (const float*)d_in[5];
  const float*     beta2 = (const float*)d_in[6];

  const int N = in_sizes[0] / IN_C;
  const int E = in_sizes[1] / 2;
  const long long* srcA = ei;
  const long long* dstA = ei + E;

  // Workspace layout (~75 MB for N=100000, E=1.6M).
  float* ws     = (float*)d_ws;
  float* hA     = ws;  ws += (size_t)N * HID_C;
  float* hB     = ws;  ws += (size_t)N * HID_C;
  float* alpha  = ws;  ws += (size_t)(E + N);
  float* inv    = ws;  ws += N;
  float* amax   = ws;  ws += N;
  float* denom  = ws;  ws += N;
  float* logits = ws;  ws += (size_t)N * OUT_C;

  const int nb  = (N + 255) / 256;                  // node grids (1 thr/node)
  const int nb4 = (N * 4 + 255) / 256;              // node grids (quad/node)
  const int ib  = (N * HID_C + 255) / 256;          // init grid
  const int eb4 = ((E + N) * 4 + 255) / 256;        // edge grids (quad/edge)
  const int gb  = (((N + 15) / 16) + 7) / 8;        // GEMM: 8 waves/block

  // h = relu(x @ W1 + b1)
  gemm_bf16x3<IN_C, 4, HID_C, true><<<gb, 256, 0, stream>>>(x, W1, b1, hA, N);
  invnorm_kernel<<<nb4, 256, 0, stream>>>(hA, inv, N);

  // prop1 (beta = 1.0)
  init_kernel<<<ib, 256, 0, stream>>>(amax, denom, hB, N);
  edge_alpha_kernel<<<eb4, 256, 0, stream>>>(hA, inv, srcA, dstA, E, N, nullptr,
                                             alpha, amax);
  edge_accum_kernel<<<eb4, 256, 0, stream>>>(hA, alpha, amax, srcA, dstA, E, N,
                                             denom, hB);
  finalize_kernel<<<nb4, 256, 0, stream>>>(hB, denom, N, inv);

  // prop2 (beta = beta2, read on device)
  init_kernel<<<ib, 256, 0, stream>>>(amax, denom, hA, N);
  edge_alpha_kernel<<<eb4, 256, 0, stream>>>(hB, inv, srcA, dstA, E, N, beta2,
                                             alpha, amax);
  edge_accum_kernel<<<eb4, 256, 0, stream>>>(hB, alpha, amax, srcA, dstA, E, N,
                                             denom, hA);
  finalize_kernel<<<nb4, 256, 0, stream>>>(hA, denom, N, nullptr);

  // logits = h @ W2 + b2  (cols padded 40 -> 48 for 3 WMMA tiles)
  gemm_bf16x3<HID_C, 3, OUT_C, false><<<gb, 256, 0, stream>>>(hA, W2, b2,
                                                              logits, N);
  logsoftmax_kernel<<<nb, 256, 0, stream>>>(logits, (float*)d_out, N);
}
